// Memory_31353261261523
// MI455X (gfx1250) — matrix operand order repaired and checked by reading, not compiled, not run on hardware
//
#include <hip/hip_runtime.h>

// ---------------------------------------------------------------------------
// MI455X (gfx1250): bf16 WMMA everywhere (v_wmma_f32_16x16x32_bf16, f32 accum).
// B fragments are staged in LDS once per block (pre-swizzled to WMMA lane
// layout) and shared by all 8 waves -> 8x less global traffic on the hot
// 327MB `p` tensor and on the conv im2col gather. gemm2 is K-split 4 ways into
// deterministic f32 partials (no atomics) + tiny reduction.
// ---------------------------------------------------------------------------

#define Bb   4
#define Tt   8
#define Cc   256
#define De   128
#define Hh   40
#define Ww   40
#define HW   1600      // 40*40
#define THW  12800     // 8*1600
#define KC   2304      // 9*256, conv contraction length
#define KCHUNKS 4      // gemm2 K-split
#define KSTEPS_PER_CHUNK (THW / KCHUNKS)   // 3200 n's = 100 ksteps

typedef __attribute__((ext_vector_type(16))) __bf16 v16bf;
typedef __attribute__((ext_vector_type(8)))  float  v8f;

union Frag {
  v16bf        v;
  __bf16       h[16];
  unsigned int w[8];
};

// A-matrix (16x32 bf16) per-lane K index for element i (this file's own order, not AMD's table):
// lanes 0-15: K0..15 ; lanes 16-31: K16..31 (contiguous, matches b_slot)
__device__ __forceinline__ int a_k(int i, int hi) {
  return i + (hi ? 16 : 0);
}

// B fragment LDS slot for matrix element (k_local 0..31, m_local 0..15):
// lane = (k>=16 ? 16:0) | m ; element i = k & 15 ; slot = lane*16 + i
__device__ __forceinline__ int b_slot(int k_local, int m_local) {
  return (((k_local >> 4) << 4) | m_local) * 16 + (k_local & 15);
}

__device__ __forceinline__ v8f wmma_bf16(const Frag& a, const Frag& b, v8f c) {
  return __builtin_amdgcn_wmma_f32_16x16x32_bf16(
      false, a.v, false, b.v, (short)0, c, false, false);
}

__device__ __forceinline__ void load_b_frag(Frag& f, const unsigned int* sB, int lane) {
#pragma unroll
  for (int j = 0; j < 8; ++j) f.w[j] = sB[lane * 8 + j];   // 32B/lane -> ds_load_b128 x2
}

// ---------------------------------------------------------------------------
// Repack conv weights (De, C, 3, 3) f32 -> bf16 [De][k], k = tap*256 + c
// ---------------------------------------------------------------------------
__global__ void cvt_w(const float* __restrict__ kw, const float* __restrict__ vw,
                      __bf16* __restrict__ wk, __bf16* __restrict__ wv) {
  int idx = blockIdx.x * blockDim.x + threadIdx.x;
  if (idx >= De * KC) return;
  int de  = idx / KC;
  int k   = idx - de * KC;
  int tap = k >> 8;       // 0..8
  int c   = k & 255;
  int src = (de * Cc + c) * 9 + tap;
  wk[idx] = (__bf16)kw[src];
  wv[idx] = (__bf16)vw[src];
}

// ---------------------------------------------------------------------------
// Implicit-GEMM 3x3 conv, pad=1. Block = 8 waves; wave w -> oc tile [16w,16w+16)
// grid.x = HW/16 spatial tiles, grid.y = frames. The im2col B tile (32k x 16px)
// is built cooperatively in LDS each kstep and shared by all 8 waves.
// mode 0 (mem frames, f = b*T + t):
//   keys[b][t*HW+hw][d] (bf16, A of gemm1), vals[b][d][t*HW+hw] (bf16, A of gemm2)
// mode 1 (query frames, f = b):
//   qk[b][d][hw] (bf16, B of gemm1), q_val f32 -> out_mem channels 128..255
// ---------------------------------------------------------------------------
__global__ void conv_wmma(const float* __restrict__ input,
                          const __bf16* __restrict__ wk, const __bf16* __restrict__ wv,
                          const float* __restrict__ bk, const float* __restrict__ bv,
                          __bf16* __restrict__ keys, __bf16* __restrict__ vals,
                          __bf16* __restrict__ qk, float* __restrict__ out_qv,
                          int mode) {
  __shared__ unsigned int sB[256];       // 512 bf16, WMMA-B lane layout

  const int f    = blockIdx.y;
  const int sp0  = blockIdx.x * 16;
  const int tid  = threadIdx.x;
  const int wave = tid >> 5;
  const int lane = tid & 31;
  const int hi   = lane >> 4;
  const int rc   = lane & 15;
  const int oc0  = wave * 16;

  const float* in = input + (size_t)f * Cc * HW;

  // builder mapping: element e -> (k_local = e>>4, m_local = e&15); e = tid, tid+256
  const unsigned short* wkrow = (const unsigned short*)wk + (size_t)(oc0 + rc) * KC;
  const unsigned short* wvrow = (const unsigned short*)wv + (size_t)(oc0 + rc) * KC;

  v8f ack = {};
  v8f acv = {};
  for (int k0 = 0; k0 < KC; k0 += 32) {
    // --- cooperative im2col build: 512 elements, coalesced rows of 16 ---
#pragma unroll
    for (int j = 0; j < 2; ++j) {
      int e = tid + j * 256;
      int kl = e >> 4, ml = e & 15;
      int k = k0 + kl;
      int tap = k >> 8;
      int c   = k & 255;
      int hw  = sp0 + ml;
      int y = hw / Ww, x = hw - (hw / Ww) * Ww;
      int yy = y + tap / 3 - 1;
      int xx = x + tap % 3 - 1;
      float val = 0.0f;
      if (yy >= 0 && yy < Hh && xx >= 0 && xx < Ww)
        val = in[(size_t)c * HW + yy * Ww + xx];
      ((__bf16*)sB)[b_slot(kl, ml)] = (__bf16)val;
    }
    __syncthreads();

    Frag Ak, Av, Bf;
    load_b_frag(Bf, sB, lane);
#pragma unroll
    for (int i = 0; i < 16; i += 2) {
      int ka = k0 + a_k(i, hi);                    // even -> dword pair load
      Ak.w[i >> 1] = *(const unsigned int*)(wkrow + ka);
      Av.w[i >> 1] = *(const unsigned int*)(wvrow + ka);
    }
    ack = wmma_bf16(Ak, Bf, ack);   // key conv
    acv = wmma_bf16(Av, Bf, acv);   // val conv (shares B fragment)
    __syncthreads();
  }

#pragma unroll
  for (int r = 0; r < 8; ++r) {
    int m  = r + hi * 8;             // D-matrix M index
    int oc = oc0 + m;
    int n  = sp0 + rc;
    float kvv = ack[r] + bk[oc];
    float vvv = acv[r] + bv[oc];
    if (mode == 0) {
      int b = f / Tt, t = f - (f / Tt) * Tt;
      size_t nn = (size_t)t * HW + n;
      keys[((size_t)b * THW + nn) * De + oc] = (__bf16)kvv;
      vals[((size_t)b * De + oc) * THW + nn] = (__bf16)vvv;
    } else {
      int b = f;
      qk[((size_t)b * De + oc) * HW + n] = (__bf16)kvv;
      out_qv[(size_t)b * (2 * De) * HW + (size_t)(De + oc) * HW + n] = vvv;
    }
  }
}

// ---------------------------------------------------------------------------
// gemm1: p[b][n][m] = (1/sqrt(128)) * sum_d keys[b][n][d] * qk[b][d][m]
// grid (HW/16, THW/128, B); wave w -> n tile blockIdx.y*8 + w.
// Whole B panel (K=128 x 16m = 4KB bf16) staged in LDS once, shared by 8 waves.
// ---------------------------------------------------------------------------
__global__ void gemm1(const __bf16* __restrict__ keys, const __bf16* __restrict__ qk,
                      float* __restrict__ p_out) {
  __shared__ unsigned int sB[1024];    // 4 ksteps x 512 bf16

  const int b    = blockIdx.z;
  const int m0   = blockIdx.x * 16;
  const int tid  = threadIdx.x;
  const int n0   = (blockIdx.y * 8 + (tid >> 5)) * 16;
  const int lane = tid & 31;
  const int hi   = lane >> 4;
  const int rc   = lane & 15;

  // stage B: qk[b][k][m0+m] for k=0..127, m=0..15 (coalesced rows of 16)
  const __bf16* qkb = qk + (size_t)b * De * HW + m0;
#pragma unroll
  for (int it = 0; it < 8; ++it) {
    int e = tid + it * 256;
    int k = e >> 4, m = e & 15;
    ((__bf16*)sB)[(k >> 5) * 512 + b_slot(k & 31, m)] = qkb[(size_t)k * HW + m];
  }
  __syncthreads();

  const unsigned short* krow = (const unsigned short*)keys + ((size_t)b * THW + n0 + rc) * De;

  v8f acc = {};
#pragma unroll
  for (int ks = 0; ks < 4; ++ks) {
    Frag A, Bf;
    load_b_frag(Bf, sB + ks * 256, lane);
#pragma unroll
    for (int i = 0; i < 16; i += 2) {
      int ka = ks * 32 + a_k(i, hi);
      A.w[i >> 1] = *(const unsigned int*)(krow + ka);
    }
    acc = wmma_bf16(A, Bf, acc);
  }

  const float scale = 0.08838834764831845f;  // 1/sqrt(128)
  float* pb = p_out + (size_t)b * THW * HW;
#pragma unroll
  for (int r = 0; r < 8; ++r) {
    int n = n0 + r + hi * 8;
    int m = m0 + rc;
    pb[(size_t)n * HW + m] = acc[r] * scale;
  }
}

// ---------------------------------------------------------------------------
// Column softmax over the memory axis (12800 elems, stride HW). One block per
// (b, m). Column staged in 51.2KB of LDS (320KB/WGP on CDNA5).
// ---------------------------------------------------------------------------
__global__ void softmax_col(float* __restrict__ p) {
  extern __shared__ float buf[];          // THW column + 256 reduction slots
  float* red = buf + THW;
  const int b = blockIdx.x / HW;
  const int m = blockIdx.x - b * HW;
  float* col = p + (size_t)b * THW * HW + m;
  const int tid = threadIdx.x;

  float mx = -3.4e38f;
  for (int n = tid; n < THW; n += 256) {
    float v = col[(size_t)n * HW];
    buf[n] = v;
    mx = fmaxf(mx, v);
  }
  red[tid] = mx;
  __syncthreads();
  for (int s = 128; s > 0; s >>= 1) {
    if (tid < s) red[tid] = fmaxf(red[tid], red[tid + s]);
    __syncthreads();
  }
  mx = red[0];
  __syncthreads();

  float sum = 0.0f;
  for (int n = tid; n < THW; n += 256) {
    float e = __expf(buf[n] - mx);
    buf[n] = e;
    sum += e;
  }
  red[tid] = sum;
  __syncthreads();
  for (int s = 128; s > 0; s >>= 1) {
    if (tid < s) red[tid] += red[tid + s];
    __syncthreads();
  }
  float inv = 1.0f / red[0];
  for (int n = tid; n < THW; n += 256) {
    col[(size_t)n * HW] = buf[n] * inv;
  }
}

// ---------------------------------------------------------------------------
// gemm2: part[chunk][b][d][m] = sum_{n in chunk} vals[b][d][n] * p[b][n][m]
// K split 4 ways (grid.y) for parallelism; deterministic partials, no atomics.
// p tile (32n x 16m f32) staged+converted to bf16 in LDS each kstep, shared by
// 8 waves -> p is streamed from HBM exactly once. Prefetch 32 rows ahead.
// ---------------------------------------------------------------------------
__global__ void gemm2(const __bf16* __restrict__ vals, const float* __restrict__ p,
                      float* __restrict__ part) {
  __shared__ unsigned int sB[256];     // 512 bf16, WMMA-B lane layout

  const int b     = blockIdx.z;
  const int chunk = blockIdx.y;
  const int m0    = blockIdx.x * 16;
  const int tid   = threadIdx.x;
  const int d0    = (tid >> 5) * 16;
  const int lane  = tid & 31;
  const int hi    = lane >> 4;
  const int rc    = lane & 15;

  const unsigned short* arow = (const unsigned short*)vals + ((size_t)b * De + d0 + rc) * THW;
  const float* pbase = p + (size_t)b * THW * HW + m0;

  const int kbeg = chunk * KSTEPS_PER_CHUNK;
  const int kend = kbeg + KSTEPS_PER_CHUNK;

  v8f acc = {};
  for (int k0 = kbeg; k0 < kend; k0 += 32) {
    // prefetch the 64B p-row segments 32 rows ahead (one lane per future row)
    if (tid < 32 && k0 + 32 + tid < kend)
      __builtin_prefetch(pbase + (size_t)(k0 + 32 + tid) * HW, 0, 0);

    // cooperative stage of p tile -> bf16 in WMMA-B layout (coalesced rows)
#pragma unroll
    for (int j = 0; j < 2; ++j) {
      int e = tid + j * 256;
      int kl = e >> 4, ml = e & 15;
      float v = pbase[(size_t)(k0 + kl) * HW + ml];
      ((__bf16*)sB)[b_slot(kl, ml)] = (__bf16)v;
    }
    __syncthreads();

    Frag A, Bf;
    load_b_frag(Bf, sB, lane);
#pragma unroll
    for (int i = 0; i < 16; i += 2) {
      int ka = k0 + a_k(i, hi);
      A.w[i >> 1] = *(const unsigned int*)(arow + ka);
    }
    acc = wmma_bf16(A, Bf, acc);
    __syncthreads();
  }

  float* dst = part + ((size_t)(chunk * Bb + b)) * De * HW;
#pragma unroll
  for (int r = 0; r < 8; ++r) {
    int d = d0 + r + hi * 8;
    int m = m0 + rc;
    dst[(size_t)d * HW + m] = acc[r];
  }
}

// Sum the 4 K-chunk partials into mem_out channels 0..127 (deterministic).
__global__ void reduce4(const float* __restrict__ part, float* __restrict__ out_mem) {
  int idx = blockIdx.x * blockDim.x + threadIdx.x;
  if (idx >= Bb * De * HW) return;
  int b = idx / (De * HW);
  int r = idx - b * De * HW;       // d*HW + m
  float s = 0.0f;
#pragma unroll
  for (int c = 0; c < KCHUNKS; ++c)
    s += part[((size_t)(c * Bb + b)) * De * HW + r];
  out_mem[(size_t)b * (2 * De) * HW + r] = s;
}

// ---------------------------------------------------------------------------
extern "C" void kernel_launch(void* const* d_in, const int* in_sizes, int n_in,
                              void* d_out, int out_size, void* d_ws, size_t ws_size,
                              hipStream_t stream) {
  (void)in_sizes; (void)n_in; (void)out_size; (void)ws_size;
  const float* mem   = (const float*)d_in[0];
  const float* query = (const float*)d_in[1];
  const float* key_w = (const float*)d_in[2];
  const float* key_b = (const float*)d_in[3];
  const float* val_w = (const float*)d_in[4];
  const float* val_b = (const float*)d_in[5];

  float* out_mem = (float*)d_out;                          // (B, 256, 40, 40)
  float* out_p   = out_mem + (size_t)Bb * 2 * De * HW;     // (B, 12800, 1600)

  // workspace: ~42 MB of staging
  __bf16* wk   = (__bf16*)d_ws;                  // [De][KC]
  __bf16* wv   = wk + (size_t)De * KC;           // [De][KC]
  __bf16* keys = wv + (size_t)De * KC;           // [B][THW][De]
  __bf16* vals = keys + (size_t)Bb * THW * De;   // [B][De][THW]
  __bf16* qk   = vals + (size_t)Bb * THW * De;   // [B][De][HW]
  float*  part = (float*)(qk + (size_t)Bb * De * HW);  // [KCHUNKS][B][De][HW]

  cvt_w<<<dim3((De * KC + 255) / 256), 256, 0, stream>>>(key_w, val_w, wk, wv);

  conv_wmma<<<dim3(HW / 16, Bb * Tt), 256, 0, stream>>>(
      mem, wk, wv, key_b, val_b, keys, vals, nullptr, nullptr, 0);
  conv_wmma<<<dim3(HW / 16, Bb), 256, 0, stream>>>(
      query, wk, wv, key_b, val_b, nullptr, nullptr, qk, out_mem, 1);

  gemm1<<<dim3(HW / 16, THW / 128, Bb), 256, 0, stream>>>(keys, qk, out_p);

  softmax_col<<<dim3(Bb * HW), 256, (THW + 256) * sizeof(float), stream>>>(out_p);

  gemm2<<<dim3(HW / 16, KCHUNKS, Bb), 256, 0, stream>>>(vals, out_p, part);

  reduce4<<<dim3((Bb * De * HW + 255) / 256), 256, 0, stream>>>(part, out_mem);
}
